// Qwen3MoeSparseExperts_21638045237560
// MI455X (gfx1250) — matrix operand order repaired and checked by reading, not compiled, not run on hardware
//
#include <hip/hip_runtime.h>

#define T_TOKENS 8192
#define H_DIM    2048
#define I_DIM    768

typedef __attribute__((ext_vector_type(16))) __bf16 v16bf;
typedef __attribute__((ext_vector_type(8)))  __bf16 v8bf;
typedef __attribute__((ext_vector_type(8)))  float  v8f;
typedef __attribute__((ext_vector_type(4)))  float  v4f;

// ---------------------------------------------------------------------------
// f32 -> bf16 pre-conversion (8 elements / thread). slice==0 => no expert
// offset (hidden states); otherwise src += (*eidx) * slice.
// Sources are read exactly once -> nontemporal loads.
// ---------------------------------------------------------------------------
__global__ __launch_bounds__(256)
void cvt_f32_bf16_kernel(const float* __restrict__ src,
                         const int*   __restrict__ eidx,
                         size_t slice,
                         unsigned short* __restrict__ dst_raw,
                         int n8) {
    __bf16* __restrict__ dst = reinterpret_cast<__bf16*>(dst_raw);
    const int i = blockIdx.x * blockDim.x + threadIdx.x;
    if (i >= n8) return;
    src += (size_t)(*eidx) * slice;
    const v4f* s = (const v4f*)src + (size_t)i * 2;
    const v4f a = __builtin_nontemporal_load(s);
    const v4f b = __builtin_nontemporal_load(s + 1);
    v8bf v;
    v[0] = (__bf16)a[0]; v[1] = (__bf16)a[1]; v[2] = (__bf16)a[2]; v[3] = (__bf16)a[3];
    v[4] = (__bf16)b[0]; v[5] = (__bf16)b[1]; v[6] = (__bf16)b[2]; v[7] = (__bf16)b[3];
    *(v8bf*)(dst + (size_t)i * 8) = v;
}

// ---------------------------------------------------------------------------
// Fragment loaders (pure bf16, no conversion)
// A 16x32: lanes 0-15 hold K = kb..kb+15 ; lanes 16-31: kb+16..kb+31.
// B 32x16: lanes 0-15 hold K = kb..kb+15 ; lanes 16-31: kb+16..kb+31.
// ---------------------------------------------------------------------------
__device__ __forceinline__ v16bf load_afrag(const __bf16* __restrict__ row,
                                            int kb, int lane) {
    const int k1 = kb + ((lane & 16) ? 16 : 0);
    const v8bf lo = *(const v8bf*)(row + k1);        // 16B
    const v8bf hi = *(const v8bf*)(row + k1 + 8);    // 16B
    v16bf v;
#pragma unroll
    for (int i = 0; i < 8; ++i) { v[i] = lo[i]; v[8 + i] = hi[i]; }
    return v;
}

__device__ __forceinline__ v16bf load_bfrag(const __bf16* __restrict__ row,
                                            int kb, int lane) {
    const int k1 = kb + ((lane & 16) ? 16 : 0);
    return *(const v16bf*)(row + k1);                // 32B (aligned)
}

// fast silu: x * sigmoid(x) with v_rcp_f32 instead of precise division
__device__ __forceinline__ float fast_silu(float x) {
    return x * __builtin_amdgcn_rcpf(1.0f + __expf(-x));
}

// ---------------------------------------------------------------------------
// Kernel 1: inter[T,I] = silu(hs @ Wg^T) * (hs @ Wu^T)   (all operands bf16)
// WG tile 128x128, 8 waves (2M x 4N), wave tile 64x32 (4M x 2N subtiles).
// ---------------------------------------------------------------------------
__global__ __launch_bounds__(256)
void moe_gateup_kernel(const unsigned short* __restrict__ hs_raw,
                       const unsigned short* __restrict__ gw_raw,
                       const unsigned short* __restrict__ uw_raw,
                       unsigned short* __restrict__ inter_raw) {
    const __bf16* __restrict__ hs = reinterpret_cast<const __bf16*>(hs_raw);
    const __bf16* __restrict__ gw = reinterpret_cast<const __bf16*>(gw_raw);
    const __bf16* __restrict__ uw = reinterpret_cast<const __bf16*>(uw_raw);
    __bf16* __restrict__ inter    = reinterpret_cast<__bf16*>(inter_raw);

    const int lane = threadIdx.x & 31;
    const int wave = threadIdx.x >> 5;
    const int wm = wave & 1;                 // 2 M-bands of 64
    const int wn = wave >> 1;                // 4 N-bands of 32
    const int rowM = blockIdx.x * 128 + wm * 64;
    const int colN = blockIdx.y * 128 + wn * 32;
    const int lr = lane & 15;

    const __bf16* arow[4];
#pragma unroll
    for (int m = 0; m < 4; ++m)
        arow[m] = hs + (size_t)(rowM + m * 16 + lr) * H_DIM;
    const __bf16* growp[2];
    const __bf16* urowp[2];
#pragma unroll
    for (int n = 0; n < 2; ++n) {
        const int nc = colN + n * 16 + lr;
        growp[n] = gw + (size_t)nc * H_DIM;
        urowp[n] = uw + (size_t)nc * H_DIM;
    }

    v8f accg[4][2] = {}, accu[4][2] = {};

    for (int kb = 0; kb < H_DIM; kb += 32) {
        v16bf a[4];
#pragma unroll
        for (int m = 0; m < 4; ++m) a[m] = load_afrag(arow[m], kb, lane);
#pragma unroll
        for (int n = 0; n < 2; ++n) {
            const v16bf bg = load_bfrag(growp[n], kb, lane);
            const v16bf bu = load_bfrag(urowp[n], kb, lane);
#pragma unroll
            for (int m = 0; m < 4; ++m) {
                accg[m][n] = __builtin_amdgcn_wmma_f32_16x16x32_bf16(
                    false, a[m], false, bg, (short)0, accg[m][n], false, false);
                accu[m][n] = __builtin_amdgcn_wmma_f32_16x16x32_bf16(
                    false, a[m], false, bu, (short)0, accu[m][n], false, false);
            }
        }
    }

    // SwiGLU epilogue, bf16 store. C/D layout: VGPR r -> row r (+8 for hi lanes).
    const int rbase = (lane & 16) ? 8 : 0;
#pragma unroll
    for (int m = 0; m < 4; ++m) {
#pragma unroll
        for (int n = 0; n < 2; ++n) {
            const int nc = colN + n * 16 + lr;
#pragma unroll
            for (int r = 0; r < 8; ++r) {
                const float x = fast_silu(accg[m][n][r]) * accu[m][n][r];
                const int row = rowM + m * 16 + rbase + r;
                inter[(size_t)row * I_DIM + nc] = (__bf16)x;
            }
        }
    }
}

// ---------------------------------------------------------------------------
// Kernel 2: out[T,H] = inter @ Wd^T   (inter, Wd bf16; out f32, streamed NT)
// WG tile 128x128, wave tile 64x32.
// ---------------------------------------------------------------------------
__global__ __launch_bounds__(256)
void moe_down_kernel(const unsigned short* __restrict__ inter_raw,
                     const unsigned short* __restrict__ dw_raw,
                     float* __restrict__ out) {
    const __bf16* __restrict__ inter = reinterpret_cast<const __bf16*>(inter_raw);
    const __bf16* __restrict__ dw    = reinterpret_cast<const __bf16*>(dw_raw);

    const int lane = threadIdx.x & 31;
    const int wave = threadIdx.x >> 5;
    const int wm = wave & 1;
    const int wn = wave >> 1;
    const int rowM = blockIdx.x * 128 + wm * 64;
    const int colN = blockIdx.y * 128 + wn * 32;
    const int lr = lane & 15;

    const __bf16* arow[4];
#pragma unroll
    for (int m = 0; m < 4; ++m)
        arow[m] = inter + (size_t)(rowM + m * 16 + lr) * I_DIM;
    const __bf16* browp[2];
#pragma unroll
    for (int n = 0; n < 2; ++n)
        browp[n] = dw + (size_t)(colN + n * 16 + lr) * I_DIM;

    v8f acc[4][2] = {};

    for (int kb = 0; kb < I_DIM; kb += 32) {
        v16bf a[4];
#pragma unroll
        for (int m = 0; m < 4; ++m) a[m] = load_afrag(arow[m], kb, lane);
#pragma unroll
        for (int n = 0; n < 2; ++n) {
            const v16bf b = load_bfrag(browp[n], kb, lane);
#pragma unroll
            for (int m = 0; m < 4; ++m) {
                acc[m][n] = __builtin_amdgcn_wmma_f32_16x16x32_bf16(
                    false, a[m], false, b, (short)0, acc[m][n], false, false);
            }
        }
    }

    // Output is write-once, never re-read: nontemporal stores keep L2 for
    // the resident hs/weights/inter working set.
    const int rbase = (lane & 16) ? 8 : 0;
#pragma unroll
    for (int m = 0; m < 4; ++m) {
#pragma unroll
        for (int n = 0; n < 2; ++n) {
            const int nc = colN + n * 16 + lr;
#pragma unroll
            for (int r = 0; r < 8; ++r) {
                const int row = rowM + m * 16 + rbase + r;
                __builtin_nontemporal_store(acc[m][n][r],
                                            out + (size_t)row * H_DIM + nc);
            }
        }
    }
}

// ---------------------------------------------------------------------------
// Launch
// ---------------------------------------------------------------------------
extern "C" void kernel_launch(void* const* d_in, const int* in_sizes, int n_in,
                              void* d_out, int out_size, void* d_ws, size_t ws_size,
                              hipStream_t stream) {
    const float* hs   = (const float*)d_in[0];
    const float* gate = (const float*)d_in[1];
    const float* up   = (const float*)d_in[2];
    const float* down = (const float*)d_in[3];
    const int*   eidx = (const int*)d_in[4];

    // Workspace layout (bf16 elements)
    const size_t HS_E  = (size_t)T_TOKENS * H_DIM;   // 16.8M
    const size_t W_E   = (size_t)I_DIM * H_DIM;      // 1.57M (gate/up/down slices)

    unsigned short* ws    = (unsigned short*)d_ws;
    unsigned short* hs_b  = ws;
    unsigned short* gw_b  = hs_b + HS_E;
    unsigned short* uw_b  = gw_b + W_E;
    unsigned short* dw_b  = uw_b + W_E;
    unsigned short* inter = dw_b + W_E;              // total ~55.6 MB

    dim3 blk(256, 1, 1);

    // Pre-convert to bf16 (hs: slice=0 disables expert offset)
    {
        const int n8_hs = (int)(HS_E / 8);
        cvt_f32_bf16_kernel<<<dim3((n8_hs + 255) / 256), blk, 0, stream>>>(
            hs, eidx, (size_t)0, hs_b, n8_hs);
        const int n8_w = (int)(W_E / 8);
        dim3 gw_grid((n8_w + 255) / 256);
        cvt_f32_bf16_kernel<<<gw_grid, blk, 0, stream>>>(gate, eidx, W_E, gw_b, n8_w);
        cvt_f32_bf16_kernel<<<gw_grid, blk, 0, stream>>>(up,   eidx, W_E, uw_b, n8_w);
        cvt_f32_bf16_kernel<<<gw_grid, blk, 0, stream>>>(down, eidx, W_E, dw_b, n8_w);
    }

    dim3 grdA(T_TOKENS / 128, I_DIM / 128, 1);        // 64 x 6
    moe_gateup_kernel<<<grdA, blk, 0, stream>>>(hs_b, gw_b, uw_b, inter);

    dim3 grdB(T_TOKENS / 128, H_DIM / 128, 1);        // 64 x 16
    moe_down_kernel<<<grdB, blk, 0, stream>>>(inter, dw_b, (float*)d_out);
}
